// LiquidNeuralNetwork_2276332666931
// MI455X (gfx1250) — matrix operand order repaired and checked by reading, not compiled, not run on hardware
//
#include <hip/hip_runtime.h>
#include <math.h>

// gfx1250 WMMA vector types
typedef __attribute__((ext_vector_type(16))) _Float16 v16h;
typedef __attribute__((ext_vector_type(8)))  float    v8f;
typedef __attribute__((ext_vector_type(2)))  _Float16 h2;

__device__ __forceinline__ float fast_tanh(float x) {
#if __has_builtin(__builtin_amdgcn_tanhf)
    return __builtin_amdgcn_tanhf(x);   // v_tanh_f32 on gfx1250
#else
    return tanhf(x);
#endif
}

// ---------------------------------------------------------------------------
// State convention (per wave, 16 batch rows):
//   val[t][v] at lane l holds element (hdim m = 16*t + v + 8*(l>=16), batch n = l&15)
// This is exactly the WMMA f32 C/D layout per 16x16 tile, 4 tiles covering 64 dims.
//
// Building a B fragment (K x 16, K=hdims) for the *next* GEMM from this layout:
//   lanes 0-15  need K = 32*kb+0..15  -> own tile 2kb (hdims +0..7) and partner
//                                        lane+16 of tile 2kb (hdims +8..15)
//   lanes 16-31 need K = 32*kb+16..31 -> partner lane-16 of tile 2kb+1 (+16..23)
//                                        and own tile 2kb+1 (+24..31)
// i.e. pack pairs to f16, xor-16 shuffle, lane-half select.
// ---------------------------------------------------------------------------
__device__ __forceinline__ void build_b(const float z[4][8], v16h b[2], bool lo) {
#pragma unroll
    for (int kb = 0; kb < 2; ++kb) {
        int p0[4], p1[4], s0[4], s1[4];
#pragma unroll
        for (int j = 0; j < 4; ++j) {
            h2 a0; a0.x = (_Float16)z[2*kb  ][2*j]; a0.y = (_Float16)z[2*kb  ][2*j+1];
            h2 a1; a1.x = (_Float16)z[2*kb+1][2*j]; a1.y = (_Float16)z[2*kb+1][2*j+1];
            p0[j] = __builtin_bit_cast(int, a0);
            p1[j] = __builtin_bit_cast(int, a1);
            s0[j] = __shfl_xor(p0[j], 16, 32);
            s1[j] = __shfl_xor(p1[j], 16, 32);
        }
        union { v16h v; int i[8]; } u;
#pragma unroll
        for (int j = 0; j < 4; ++j) {
            u.i[j]     = lo ? p0[j] : s1[j];
            u.i[4 + j] = lo ? s0[j] : p1[j];
        }
        b[kb] = u.v;
    }
}

// D_t = A(W)_t,kb x B_kb + bias_t  over kb=0..1 ; W fragments pre-swizzled in LDS.
__device__ __forceinline__ void gemm64(const _Float16* __restrict__ wA,
                                       const float* __restrict__ bias,
                                       const v16h b[2], float out[4][8],
                                       int m_lo, int hi16) {
#pragma unroll
    for (int t = 0; t < 4; ++t) {
        union { v8f v; float4 f[2]; } c;
        const float4* bp = (const float4*)(bias + 16*t + 8*hi16); // 32B aligned
        c.f[0] = bp[0];
        c.f[1] = bp[1];
#pragma unroll
        for (int kb = 0; kb < 2; ++kb) {
            union { v16h v; int4 q[2]; } a;
            const char* fb = (const char*)wA + (t*2 + kb)*1024 + m_lo*64 + hi16*32;
            a.q[0] = *(const int4*)(fb);        // k = 16*hi16 + 0..7
            a.q[1] = *(const int4*)(fb + 16);   // k = 16*hi16 + 8..15
            c.v = __builtin_amdgcn_wmma_f32_16x16x32_f16(
                false, a.v, false, b[kb], (short)0, c.v, false, false);
        }
#pragma unroll
        for (int v = 0; v < 8; ++v) out[t][v] = c.v[v];
    }
}

// f(z) = tanh(z@W1+b1)@W2+b2, fully transposed/register-resident.
__device__ __forceinline__ void feval(const float z[4][8], float out[4][8],
                                      const _Float16* w1A, const _Float16* w2A,
                                      const float* b1L, const float* b2L,
                                      int m_lo, int hi16, bool lo) {
    v16h B[2];
    build_b(z, B, lo);
    float h[4][8];
    gemm64(w1A, b1L, B, h, m_lo, hi16);
#pragma unroll
    for (int t = 0; t < 4; ++t)
#pragma unroll
        for (int v = 0; v < 8; ++v) h[t][v] = fast_tanh(h[t][v]);
    build_b(h, B, lo);
    gemm64(w2A, b2L, B, out, m_lo, hi16);
}

__global__ __launch_bounds__(256)
void lnn_dopri5_kernel(const float* __restrict__ x,    const float* __restrict__ ts,
                       const float* __restrict__ Win,  const float* __restrict__ bin,
                       const float* __restrict__ W1,   const float* __restrict__ b1g,
                       const float* __restrict__ W2,   const float* __restrict__ b2g,
                       const float* __restrict__ Wout, const float* __restrict__ boutg,
                       float* __restrict__ out, int Bn) {
    __shared__ alignas(16) _Float16 w1A[4096];   // 8 frags x (16 rows x 32 f16)
    __shared__ alignas(16) _Float16 w2A[4096];
    __shared__ alignas(16) _Float16 winA[2048];  // 4 frags (kb=0 only, K padded to 32)
    __shared__ alignas(16) float b_inL[64], b1L[64], b2L[64];
    __shared__ alignas(16) float woutL[256];     // [m][4], j<3 valid

    const int tid = threadIdx.x;

    // ---- one-time weight swizzle into LDS (A-fragment layout) ----
    for (int idx = tid; idx < 4096; idx += 256) {
        int frag = idx >> 9, e = idx & 511, m = e >> 5, k = e & 31;
        int t = frag >> 1, kb = frag & 1;
        int src = (kb*32 + k)*64 + (t*16 + m);   // W[k_global][m_global]
        w1A[idx] = (_Float16)W1[src];
        w2A[idx] = (_Float16)W2[src];
    }
    for (int idx = tid; idx < 2048; idx += 256) {
        int t = idx >> 9, e = idx & 511, m = e >> 5, k = e & 31;
        winA[idx] = (k < 14) ? (_Float16)Win[k*64 + t*16 + m] : (_Float16)0.0f;
    }
    if (tid < 64) { b_inL[tid] = bin[tid]; b1L[tid] = b1g[tid]; b2L[tid] = b2g[tid]; }
    for (int idx = tid; idx < 256; idx += 256) {
        int m = idx >> 2, j = idx & 3;
        woutL[idx] = (j < 3) ? Wout[m*3 + j] : 0.0f;
    }
    __syncthreads();

    const int lane  = tid & 31;
    const int wave  = tid >> 5;
    const int m_lo  = lane & 15;
    const int hi16  = lane >> 4;
    const bool lo   = (hi16 == 0);
    const int rowbase = (blockIdx.x * 8 + wave) * 16;
    const int row     = rowbase + m_lo;

    // ---- y0 = x @ W_in + b_in  via one WMMA per tile (K=14 padded to 32) ----
    union { v16h v; int i[8]; } xb;
#pragma unroll
    for (int j = 0; j < 8; ++j) xb.i[j] = 0;
    if (lo && row < Bn) {
        const float* xr = x + (size_t)row * 14;
        float xv[16];
#pragma unroll
        for (int k = 0; k < 14; ++k) xv[k] = xr[k];
        xv[14] = 0.0f; xv[15] = 0.0f;
#pragma unroll
        for (int j = 0; j < 8; ++j) {
            h2 p; p.x = (_Float16)xv[2*j]; p.y = (_Float16)xv[2*j + 1];
            xb.i[j] = __builtin_bit_cast(int, p);
        }
    }

    float y[4][8];
#pragma unroll
    for (int t = 0; t < 4; ++t) {
        union { v8f v; float4 f[2]; } c;
        const float4* bp = (const float4*)(b_inL + 16*t + 8*hi16);
        c.f[0] = bp[0]; c.f[1] = bp[1];
        union { v16h v; int4 q[2]; } a;
        const char* fb = (const char*)winA + t*1024 + m_lo*64 + hi16*32;
        a.q[0] = *(const int4*)(fb);
        a.q[1] = *(const int4*)(fb + 16);
        c.v = __builtin_amdgcn_wmma_f32_16x16x32_f16(
            false, a.v, false, xb.v, (short)0, c.v, false, false);
#pragma unroll
        for (int v = 0; v < 8; ++v) y[t][v] = c.v[v];
    }

    // ---- DOPRI5 fixed-step integration (dt * tableau precomputed, uniform) ----
    const float dt  = (ts[1] - ts[0]) * (1.0f / 32.0f);
    const float c21 = dt * (float)(1.0/5.0);
    const float c31 = dt * (float)(3.0/40.0),      c32 = dt * (float)(9.0/40.0);
    const float c41 = dt * (float)(44.0/45.0),     c42 = dt * (float)(-56.0/15.0);
    const float c43 = dt * (float)(32.0/9.0);
    const float c51 = dt * (float)(19372.0/6561.0), c52 = dt * (float)(-25360.0/2187.0);
    const float c53 = dt * (float)(64448.0/6561.0), c54 = dt * (float)(-212.0/729.0);
    const float c61 = dt * (float)(9017.0/3168.0),  c62 = dt * (float)(-355.0/33.0);
    const float c63 = dt * (float)(46732.0/5247.0), c64 = dt * (float)(49.0/176.0);
    const float c65 = dt * (float)(-5103.0/18656.0);
    const float d1  = dt * (float)(35.0/384.0),     d3  = dt * (float)(500.0/1113.0);
    const float d4  = dt * (float)(125.0/192.0),    d5  = dt * (float)(-2187.0/6784.0);
    const float d6  = dt * (float)(11.0/84.0);

#pragma unroll 1
    for (int s = 0; s < 32; ++s) {
        float k1[4][8], k2[4][8], k3[4][8], k4[4][8], k5[4][8], k6[4][8], z[4][8];
        feval(y, k1, w1A, w2A, b1L, b2L, m_lo, hi16, lo);
#pragma unroll
        for (int t = 0; t < 4; ++t)
#pragma unroll
            for (int v = 0; v < 8; ++v) z[t][v] = y[t][v] + c21*k1[t][v];
        feval(z, k2, w1A, w2A, b1L, b2L, m_lo, hi16, lo);
#pragma unroll
        for (int t = 0; t < 4; ++t)
#pragma unroll
            for (int v = 0; v < 8; ++v) z[t][v] = y[t][v] + c31*k1[t][v] + c32*k2[t][v];
        feval(z, k3, w1A, w2A, b1L, b2L, m_lo, hi16, lo);
#pragma unroll
        for (int t = 0; t < 4; ++t)
#pragma unroll
            for (int v = 0; v < 8; ++v)
                z[t][v] = y[t][v] + c41*k1[t][v] + c42*k2[t][v] + c43*k3[t][v];
        feval(z, k4, w1A, w2A, b1L, b2L, m_lo, hi16, lo);
#pragma unroll
        for (int t = 0; t < 4; ++t)
#pragma unroll
            for (int v = 0; v < 8; ++v)
                z[t][v] = y[t][v] + c51*k1[t][v] + c52*k2[t][v] + c53*k3[t][v] + c54*k4[t][v];
        feval(z, k5, w1A, w2A, b1L, b2L, m_lo, hi16, lo);
#pragma unroll
        for (int t = 0; t < 4; ++t)
#pragma unroll
            for (int v = 0; v < 8; ++v)
                z[t][v] = y[t][v] + c61*k1[t][v] + c62*k2[t][v] + c63*k3[t][v]
                                  + c64*k4[t][v] + c65*k5[t][v];
        feval(z, k6, w1A, w2A, b1L, b2L, m_lo, hi16, lo);
#pragma unroll
        for (int t = 0; t < 4; ++t)
#pragma unroll
            for (int v = 0; v < 8; ++v)
                y[t][v] = y[t][v] + d1*k1[t][v] + d3*k3[t][v] + d4*k4[t][v]
                                  + d5*k5[t][v] + d6*k6[t][v];
    }

    // ---- output head: out[n][j] = sum_m y[n][m]*Wout[m][j] + bout[j] ----
    float a0 = 0.0f, a1 = 0.0f, a2 = 0.0f;
#pragma unroll
    for (int t = 0; t < 4; ++t)
#pragma unroll
        for (int v = 0; v < 8; ++v) {
            float4 w = *((const float4*)woutL + (16*t + v + 8*hi16));
            float yv = y[t][v];
            a0 += w.x * yv; a1 += w.y * yv; a2 += w.z * yv;
        }
    // lanes l and l+16 hold complementary m's of the same batch row -> xor-16 reduce
    a0 += __shfl_xor(a0, 16, 32);
    a1 += __shfl_xor(a1, 16, 32);
    a2 += __shfl_xor(a2, 16, 32);

    if (lo && row < Bn) {
        out[(size_t)row*3 + 0] = a0 + boutg[0];
        out[(size_t)row*3 + 1] = a1 + boutg[1];
        out[(size_t)row*3 + 2] = a2 + boutg[2];
    }
}

extern "C" void kernel_launch(void* const* d_in, const int* in_sizes, int n_in,
                              void* d_out, int out_size, void* d_ws, size_t ws_size,
                              hipStream_t stream) {
    (void)n_in; (void)d_ws; (void)ws_size; (void)out_size;
    const float* x    = (const float*)d_in[0];
    const float* ts   = (const float*)d_in[1];
    const float* Win  = (const float*)d_in[2];
    const float* bin  = (const float*)d_in[3];
    const float* W1   = (const float*)d_in[4];
    const float* b1   = (const float*)d_in[5];
    const float* W2   = (const float*)d_in[6];
    const float* b2   = (const float*)d_in[7];
    const float* Wout = (const float*)d_in[8];
    const float* bout = (const float*)d_in[9];
    float* out = (float*)d_out;

    const int B = in_sizes[0] / 14;           // 131072
    const int rows_per_block = 8 * 16;        // 8 waves x 16 rows
    const int blocks = (B + rows_per_block - 1) / rows_per_block;

    lnn_dopri5_kernel<<<blocks, 256, 0, stream>>>(
        x, ts, Win, bin, W1, b1, W2, b2, Wout, bout, out, B);
}
